// prcbm_60945585930925
// MI455X (gfx1250) — compile-verified, matrix operand order checked by reading, not run on hardware
//
#include <hip/hip_runtime.h>
#include <hip/hip_bf16.h>
#include <math.h>

// ---------------- problem constants ----------------
#define NL   12      // layers
#define NB   32      // batch
#define NS   197     // seq
#define DV   768     // vit dim
#define DD   512     // model dim
#define NCPT 48      // concepts
#define VR   49      // vt rows (cls + pooled)
#define NROW (NB * NCPT)   // 1536 recurrent rows

typedef __attribute__((ext_vector_type(16))) __bf16 v16bf;
typedef __attribute__((ext_vector_type(8)))  float  v8f;

union Frag {
    v16bf v;
    unsigned int   w[8];
    unsigned short h[16];
};

// Native bf16 conversion (RNE) -> raw bits.
__device__ __forceinline__ unsigned short f2bf(float f) {
    __bf16 b = (__bf16)f;
    return __builtin_bit_cast(unsigned short, b);
}

// Per-v fragment dword offset (compile-time constant after unroll):
// element k-offset within a 32-wide K step is (v&3)*2 + (v>>2)*16 (+ hf*8 hoisted
// into the base pointer), i.e. dword offset (v&3) + (v>>2)*8.
__device__ __forceinline__ constexpr int fragc(int v) {
    return (v & 3) + ((v >> 2) << 3);
}

// ---------------- stage 0: pooling + bf16 staging ----------------
// Builds X (NL*NB*49, 768) bf16:  row i==0 -> cls token, rows 1..48 -> adaptive avg pool.
// Also extracts cls tokens (NL*NB, 768) bf16. Single coalesced pass over feats (232 MB),
// which is the HBM roofline floor for this problem (~10 us at 23.3 TB/s).
__global__ void pool_kernel(const float* __restrict__ feats,
                            unsigned short* __restrict__ Xbf,
                            unsigned short* __restrict__ clsbf) {
    int row = blockIdx.x;                 // (l*NB+b)*49 + i
    int i   = row % VR;
    int lb  = row / VR;
    const float* fbase = feats + (size_t)lb * NS * DV;
    unsigned short* orow = Xbf + (size_t)row * DV;
    if (i == 0) {
        for (int d = threadIdx.x; d < DV; d += blockDim.x) {
            unsigned short bv = f2bf(fbase[d]);
            orow[d] = bv;
            clsbf[(size_t)lb * DV + d] = bv;
        }
    } else {
        int c = i - 1;
        int s = (c * 196) / NCPT;
        int e = ((c + 1) * 196 + NCPT - 1) / NCPT;
        float inv = 1.0f / (float)(e - s);
        for (int d = threadIdx.x; d < DV; d += blockDim.x) {
            float acc = 0.f;
            for (int t = s; t < e; ++t) acc += fbase[(size_t)(1 + t) * DV + d];
            orow[d] = f2bf(acc * inv);
        }
    }
}

// Transpose a (512, K) fp32 weight into PAIR-INTERLEAVED B-layout bf16:
//   dst[(k>>1)*1024 + 2*n + (k&1)] = src[n*K + k]
// so a WMMA B-fragment k-pair at column n is ONE aligned dword.
__global__ void wtrans_kernel(const float* __restrict__ src,
                              unsigned short* __restrict__ dst, int K) {
    int idx = blockIdx.x * blockDim.x + threadIdx.x;  // k*512 + n
    int n = idx & (DD - 1);
    int k = idx >> 9;
    if (k < K)
        dst[(size_t)(k >> 1) * (2 * DD) + 2 * n + (k & 1)] = f2bf(src[(size_t)n * K + k]);
}

__global__ void initq_kernel(const float* __restrict__ ce,
                             float* __restrict__ qF, unsigned short* __restrict__ qB) {
    int idx = blockIdx.x * blockDim.x + threadIdx.x;  // row*512 + n
    int n   = idx & (DD - 1);
    int row = idx >> 9;
    float v = ce[(row % NCPT) * DD + n];
    qF[idx] = v;
    qB[idx] = f2bf(v);
}

__global__ void zero_kernel(float* p) { *p = 0.f; }

// ---------------- generic bf16 WMMA GEMM: O = A @ B (+bias), bf16 out ----------------
// A: row-major (M,K) bf16, lda; B: pair-interleaved (K/2, 512, 2) bf16; O: (M,512) bf16.
// 8 waves/block, one 16x16 tile per wave. Fragment loads are dwords with constant
// immediate offsets from a single per-iteration base.
__global__ void gemm_bf16_kernel(const unsigned short* __restrict__ A, int lda,
                                 const unsigned short* __restrict__ Bp,
                                 const float* __restrict__ bias,
                                 unsigned short* __restrict__ Obf,
                                 int Ntiles, int K) {
    int wave = threadIdx.x >> 5, lane = threadIdx.x & 31;
    int tid  = blockIdx.x * 8 + wave;
    int mt = tid / Ntiles, nt = tid % Ntiles;
    int hf = lane >> 4, l15 = lane & 15;
    int mrow = mt * 16 + l15;
    int ncol = nt * 16 + l15;
    const unsigned int* ap0 = (const unsigned int*)(A + (size_t)mrow * lda + hf * 8);
    const unsigned int* bp0 = (const unsigned int*)Bp + ncol + hf * 2048;
    v8f acc = {};
    for (int k0 = 0; k0 < K; k0 += 32) {
        const unsigned int* ap = ap0 + (k0 >> 1);
        const unsigned int* bp = bp0 + (size_t)(k0 >> 1) * 512;
        Frag a, b;
#pragma unroll
        for (int v = 0; v < 8; ++v) {
            a.w[v] = ap[fragc(v)];
            b.w[v] = bp[fragc(v) * 512];
        }
        acc = __builtin_amdgcn_wmma_f32_16x16x32_bf16(false, a.v, false, b.v,
                                                      (short)0, acc, false, false);
    }
    float bi = bias ? bias[ncol] : 0.f;
#pragma unroll
    for (int v = 0; v < 8; ++v) {
        int m = mt * 16 + v + hf * 8;
        Obf[(size_t)m * DD + ncol] = f2bf(acc[v] + bi);
    }
}

// ---------------- per-layer: attention scores + softmax ----------------
// One block per batch b. 12 waves: 3 M-tiles x 4 N-tiles of the 48x64 score matrix.
// B operand is vt^T, gathered from contiguous vt rows (dwords). Padded columns
// n>=49 are clamped to row 48 (finite garbage, masked out by the softmax).
__global__ void attn_kernel(const unsigned short* __restrict__ qB,   // (1536,512)
                            const unsigned short* __restrict__ vtl,  // (NB,49,512)
                            unsigned short* __restrict__ attnB) {    // (1536,64)
    __shared__ float sc[NCPT * 64];
    int b = blockIdx.x;
    int wave = threadIdx.x >> 5, lane = threadIdx.x & 31;
    int hf = lane >> 4, l15 = lane & 15;
    int mt = wave >> 2, nt = wave & 3;
    int m = mt * 16 + l15;                 // 0..47
    int n = nt * 16 + l15;                 // 0..63
    int nv = (n < VR) ? n : (VR - 1);
    const unsigned int* ap0 = (const unsigned int*)(qB + (size_t)(b * NCPT + m) * DD + hf * 8);
    const unsigned int* bp0 = (const unsigned int*)(vtl + (size_t)(b * VR + nv) * DD + hf * 8);
    v8f acc = {};
    for (int k0 = 0; k0 < DD; k0 += 32) {
        const unsigned int* ap = ap0 + (k0 >> 1);
        const unsigned int* bp = bp0 + (k0 >> 1);
        Frag a, bb;
#pragma unroll
        for (int v = 0; v < 8; ++v) {
            a.w[v]  = ap[fragc(v)];
            bb.w[v] = bp[fragc(v)];
        }
        acc = __builtin_amdgcn_wmma_f32_16x16x32_bf16(false, a.v, false, bb.v,
                                                      (short)0, acc, false, false);
    }
#pragma unroll
    for (int v = 0; v < 8; ++v)
        sc[(mt * 16 + v + hf * 8) * 64 + n] = acc[v];
    __syncthreads();
    int r = threadIdx.x;
    if (r < NCPT) {
        float mx = -1e30f;
        for (int c = 0; c < VR; ++c) mx = fmaxf(mx, sc[r * 64 + c]);
        float sum = 0.f;
        for (int c = 0; c < VR; ++c) {
            float ev = __expf(sc[r * 64 + c] - mx);
            sc[r * 64 + c] = ev;
            sum += ev;
        }
        float inv = 1.0f / sum;
        unsigned short* orow = attnB + (size_t)(b * NCPT + r) * 64;
        for (int c = 0; c < VR; ++c) orow[c] = f2bf(sc[r * 64 + c] * inv);
        for (int c = VR; c < 64; ++c) orow[c] = 0;
    }
}

// ---------------- per-layer: fuse = attn @ vt ----------------
__global__ void fuse_kernel(const unsigned short* __restrict__ attnB, // (1536,64)
                            const unsigned short* __restrict__ vtl,   // (NB,49,512)
                            float* __restrict__ fuseF,
                            unsigned short* __restrict__ fuseB) {
    int wave = threadIdx.x >> 5, lane = threadIdx.x & 31;
    int tid  = blockIdx.x * 8 + wave;       // 0..3071
    int b = tid / 96, rem = tid % 96;
    int mt = rem / 32, nt = rem % 32;
    int hf = lane >> 4, l15 = lane & 15;
    int m = mt * 16 + l15;
    int n = nt * 16 + l15;
    const unsigned int* ap0 = (const unsigned int*)(attnB + (size_t)(b * NCPT + m) * 64 + hf * 8);
    v8f acc = {};
    for (int k0 = 0; k0 < 64; k0 += 32) {
        const unsigned int* ap = ap0 + (k0 >> 1);
        Frag a, bb;
#pragma unroll
        for (int v = 0; v < 8; ++v) {
            a.w[v] = ap[fragc(v)];
            int kk = k0 + ((v & 3) << 1) + (hf << 3) + ((v >> 2) << 4);
            bb.h[2*v]   = (kk     < VR) ? vtl[(size_t)(b * VR + kk    ) * DD + n] : (unsigned short)0;
            bb.h[2*v+1] = (kk + 1 < VR) ? vtl[(size_t)(b * VR + kk + 1) * DD + n] : (unsigned short)0;
        }
        acc = __builtin_amdgcn_wmma_f32_16x16x32_bf16(false, a.v, false, bb.v,
                                                      (short)0, acc, false, false);
    }
#pragma unroll
    for (int v = 0; v < 8; ++v) {
        int mm = mt * 16 + v + hf * 8;
        size_t idx = (size_t)(b * NCPT + mm) * DD + n;
        fuseF[idx] = acc[v];
        fuseB[idx] = f2bf(acc[v]);
    }
}

// ---------------- per-layer: consistency loss term ----------------
__global__ void loss_kernel(const float* __restrict__ a, const float* __restrict__ b,
                            float* __restrict__ acc) {
    __shared__ float red[256];
    int i = blockIdx.x * blockDim.x + threadIdx.x;
    float d = a[i] - b[i];
    red[threadIdx.x] = d * d;
    __syncthreads();
    for (int s = 128; s > 0; s >>= 1) {
        if (threadIdx.x < s) red[threadIdx.x] += red[threadIdx.x + s];
        __syncthreads();
    }
    if (threadIdx.x == 0)
        atomicAdd(acc, red[0] * (1.0f / ((float)NROW * (float)DD)));
}

// ---------------- per-layer: gate/upd GEMMs + gated update ----------------
// gate = sigmoid([fuse|q|cls] @ Wg^T + bg), upd = tanh([fuse|q] @ Wu^T + bu),
// q' = g*q + (1-g)*u.  M=1536, N=512, Kg=1536, Ku=1024; both weights pair-interleaved.
__global__ void gateupd_kernel(const unsigned short* __restrict__ fuseB, // (1536,512)
                               const unsigned short* __restrict__ qBin,  // (1536,512)
                               const unsigned short* __restrict__ clsl,  // (NB,512)
                               const unsigned short* __restrict__ WgP,   // pair (768,512,2)
                               const unsigned short* __restrict__ WuP,   // pair (512,512,2)
                               const float* __restrict__ bg,
                               const float* __restrict__ bu,
                               const float* __restrict__ qFin,
                               float* __restrict__ qFout,
                               unsigned short* __restrict__ qBout) {
    int wave = threadIdx.x >> 5, lane = threadIdx.x & 31;
    int tid  = blockIdx.x * 8 + wave;       // 0..3071
    int mt = tid >> 5, nt = tid & 31;
    int hf = lane >> 4, l15 = lane & 15;
    int m = mt * 16 + l15;
    int n = nt * 16 + l15;
    const unsigned int* fp0 = (const unsigned int*)(fuseB + (size_t)m * DD + hf * 8);
    const unsigned int* qp0 = (const unsigned int*)(qBin + (size_t)m * DD + hf * 8);
    const unsigned int* cp0 = (const unsigned int*)(clsl + (size_t)(m / NCPT) * DD + hf * 8);
    const unsigned int* gp0 = (const unsigned int*)WgP + n + hf * 2048;
    const unsigned int* up0 = (const unsigned int*)WuP + n + hf * 2048;
    v8f aG = {}, aU = {};
    // phase 1: k in [0,512)  — A = fuse
    for (int k0 = 0; k0 < DD; k0 += 32) {
        const unsigned int* ap = fp0 + (k0 >> 1);
        const unsigned int* gp = gp0 + (size_t)(k0 >> 1) * 512;
        const unsigned int* up = up0 + (size_t)(k0 >> 1) * 512;
        Frag a, bG, bU;
#pragma unroll
        for (int v = 0; v < 8; ++v) {
            a.w[v]  = ap[fragc(v)];
            bG.w[v] = gp[fragc(v) * 512];
            bU.w[v] = up[fragc(v) * 512];
        }
        aG = __builtin_amdgcn_wmma_f32_16x16x32_bf16(false, a.v, false, bG.v, (short)0, aG, false, false);
        aU = __builtin_amdgcn_wmma_f32_16x16x32_bf16(false, a.v, false, bU.v, (short)0, aU, false, false);
    }
    // phase 2: k in [512,1024) — A = q
    for (int k0 = 0; k0 < DD; k0 += 32) {
        const unsigned int* ap = qp0 + (k0 >> 1);
        const unsigned int* gp = gp0 + (size_t)((DD + k0) >> 1) * 512;
        const unsigned int* up = up0 + (size_t)((DD + k0) >> 1) * 512;
        Frag a, bG, bU;
#pragma unroll
        for (int v = 0; v < 8; ++v) {
            a.w[v]  = ap[fragc(v)];
            bG.w[v] = gp[fragc(v) * 512];
            bU.w[v] = up[fragc(v) * 512];
        }
        aG = __builtin_amdgcn_wmma_f32_16x16x32_bf16(false, a.v, false, bG.v, (short)0, aG, false, false);
        aU = __builtin_amdgcn_wmma_f32_16x16x32_bf16(false, a.v, false, bU.v, (short)0, aU, false, false);
    }
    // phase 3: k in [1024,1536) — A = cls_ctx (gate only)
    for (int k0 = 0; k0 < DD; k0 += 32) {
        const unsigned int* ap = cp0 + (k0 >> 1);
        const unsigned int* gp = gp0 + (size_t)((2 * DD + k0) >> 1) * 512;
        Frag a, bG;
#pragma unroll
        for (int v = 0; v < 8; ++v) {
            a.w[v]  = ap[fragc(v)];
            bG.w[v] = gp[fragc(v) * 512];
        }
        aG = __builtin_amdgcn_wmma_f32_16x16x32_bf16(false, a.v, false, bG.v, (short)0, aG, false, false);
    }
    float bgn = bg[n], bun = bu[n];
#pragma unroll
    for (int v = 0; v < 8; ++v) {
        int mm = mt * 16 + v + hf * 8;
        size_t idx = (size_t)mm * DD + n;
        float g = 1.0f / (1.0f + __expf(-(aG[v] + bgn)));
        float u = tanhf(aU[v] + bun);
        float qn = g * qFin[idx] + (1.0f - g) * u;
        qFout[idx] = qn;
        qBout[idx] = f2bf(qn);
    }
}

// ---------------- epilogue: cosine logits + classifier + loss write ----------------
__global__ void final_kernel(const float* __restrict__ last,  // (1536,512)
                             const float* __restrict__ ce,    // (48,512), pre-normalized
                             const float* __restrict__ Wcls, const float* __restrict__ bcls,
                             const float* __restrict__ lossAcc,
                             float* __restrict__ out) {
    __shared__ float invn[NCPT];
    __shared__ float dots[288];
    __shared__ float il[NCPT];
    int b = blockIdx.x;
    int wave = threadIdx.x >> 5, lane = threadIdx.x & 31;
    for (int r = wave; r < NCPT; r += 8) {
        const float* row = last + (size_t)(b * NCPT + r) * DD;
        float s = 0.f;
        for (int j = lane; j < DD; j += 32) { float x = row[j]; s += x * x; }
        for (int o = 16; o > 0; o >>= 1) s += __shfl_xor(s, o, 32);
        if (lane == 0) invn[r] = 1.0f / fmaxf(sqrtf(s), 1e-12f);
    }
    __syncthreads();
    for (int idx = wave; idx < 288; idx += 8) {
        int k = idx / 36, rm = idx % 36, p = rm / 6, qi = rm % 6;
        const float* ra = last + (size_t)(b * NCPT + k * 6 + p) * DD;
        const float* rb = ce + (size_t)(k * 6 + qi) * DD;
        float s = 0.f;
        for (int j = lane; j < DD; j += 32) s += ra[j] * rb[j];
        for (int o = 16; o > 0; o >>= 1) s += __shfl_xor(s, o, 32);
        if (lane == 0) dots[idx] = s * invn[k * 6 + p];
    }
    __syncthreads();
    if (threadIdx.x < NCPT) {
        int j = threadIdx.x, k = j / 6, qi = j % 6;
        float s = 0.f;
        for (int p = 0; p < 6; ++p) s += dots[k * 36 + p * 6 + qi];
        float v = 100.0f * s * (1.0f / 6.0f);
        il[j] = v;
        out[NB * 7 + b * NCPT + j] = v;
    }
    __syncthreads();
    if (threadIdx.x < 7) {
        float s = bcls[threadIdx.x];
        for (int j = 0; j < NCPT; ++j) s += il[j] * Wcls[threadIdx.x * NCPT + j];
        out[b * 7 + threadIdx.x] = s;
    }
    if (b == 0 && threadIdx.x == 0) out[NB * 7 + NB * NCPT] = *lossAcc;
}

// ---------------- host launcher ----------------
extern "C" void kernel_launch(void* const* d_in, const int* in_sizes, int n_in,
                              void* d_out, int out_size, void* d_ws, size_t ws_size,
                              hipStream_t stream) {
    const float* feats  = (const float*)d_in[0];
    const float* ce     = (const float*)d_in[1];
    const float* W_lin  = (const float*)d_in[2];
    const float* b_lin  = (const float*)d_in[3];
    const float* W_proj = (const float*)d_in[4];
    const float* W_gate = (const float*)d_in[5];
    const float* b_gate = (const float*)d_in[6];
    const float* W_upd  = (const float*)d_in[7];
    const float* b_upd  = (const float*)d_in[8];
    const float* W_cls  = (const float*)d_in[9];
    const float* b_cls  = (const float*)d_in[10];
    float* out = (float*)d_out;

    char* ws = (char*)d_ws;
    size_t off = 0;
    auto carve = [&](size_t bytes) -> void* {
        void* p = ws + off;
        off = (off + bytes + 255) & ~(size_t)255;
        return p;
    };
    const size_t MROWS = (size_t)NL * NB * VR;   // 18816
    unsigned short* Xbf     = (unsigned short*)carve(MROWS * DV * 2);
    unsigned short* clsbf   = (unsigned short*)carve((size_t)NL * NB * DV * 2);
    unsigned short* WlinP   = (unsigned short*)carve((size_t)DV * DD * 2);
    unsigned short* WprojP  = (unsigned short*)carve((size_t)DV * DD * 2);
    unsigned short* WgP     = (unsigned short*)carve((size_t)3 * DD * DD * 2);
    unsigned short* WuP     = (unsigned short*)carve((size_t)2 * DD * DD * 2);
    unsigned short* vtbf    = (unsigned short*)carve(MROWS * DD * 2);
    unsigned short* clsctx  = (unsigned short*)carve((size_t)NL * NB * DD * 2);
    unsigned short* attnB   = (unsigned short*)carve((size_t)NROW * 64 * 2);
    float*          qF[2]   = { (float*)carve((size_t)NROW * DD * 4),
                                (float*)carve((size_t)NROW * DD * 4) };
    unsigned short* qB[2]   = { (unsigned short*)carve((size_t)NROW * DD * 2),
                                (unsigned short*)carve((size_t)NROW * DD * 2) };
    float*          fuseF[2]= { (float*)carve((size_t)NROW * DD * 4),
                                (float*)carve((size_t)NROW * DD * 4) };
    unsigned short* fuseB[2]= { (unsigned short*)carve((size_t)NROW * DD * 2),
                                (unsigned short*)carve((size_t)NROW * DD * 2) };
    float* lossAcc = (float*)carve(256);

    zero_kernel<<<1, 1, 0, stream>>>(lossAcc);
    pool_kernel<<<(int)MROWS, 256, 0, stream>>>(feats, Xbf, clsbf);
    wtrans_kernel<<<(DV * DD) / 256, 256, 0, stream>>>(W_lin,  WlinP, DV);
    wtrans_kernel<<<(DV * DD) / 256, 256, 0, stream>>>(W_proj, WprojP, DV);
    wtrans_kernel<<<(3 * DD * DD) / 256, 256, 0, stream>>>(W_gate, WgP, 3 * DD);
    wtrans_kernel<<<(2 * DD * DD) / 256, 256, 0, stream>>>(W_upd,  WuP, 2 * DD);
    initq_kernel<<<(NROW * DD) / 256, 256, 0, stream>>>(ce, qF[0], qB[0]);

    // All-layer vt GEMM: (18816,768) @ (768,512) + bias -> bf16. 1176x32 tiles.
    gemm_bf16_kernel<<<(1176 * 32) / 8, 256, 0, stream>>>(Xbf, DV, WlinP, b_lin, vtbf, 32, DV);
    // All-layer cls_ctx GEMM: (384,768) @ (768,512) -> bf16. 24x32 tiles.
    gemm_bf16_kernel<<<(24 * 32) / 8, 256, 0, stream>>>(clsbf, DV, WprojP, nullptr, clsctx, 32, DV);

    for (int l = 0; l < NL; ++l) {
        const unsigned short* vtl = vtbf + (size_t)l * NB * VR * DD;
        const unsigned short* cll = clsctx + (size_t)l * NB * DD;
        int pi = l & 1, po = 1 - pi;      // q double-buffer
        int fp = l & 1;                   // fuse double-buffer
        attn_kernel<<<NB, 384, 0, stream>>>(qB[pi], vtl, attnB);
        fuse_kernel<<<(NB * 96) / 8, 256, 0, stream>>>(attnB, vtl, fuseF[fp], fuseB[fp]);
        if (l > 0)
            loss_kernel<<<(NROW * DD) / 256, 256, 0, stream>>>(fuseF[fp], fuseF[1 - fp], lossAcc);
        gateupd_kernel<<<(96 * 32) / 8, 256, 0, stream>>>(fuseB[fp], qB[pi], cll, WgP, WuP,
                                                          b_gate, b_upd, qF[pi], qF[po], qB[po]);
    }
    final_kernel<<<NB, 256, 0, stream>>>(fuseF[(NL - 1) & 1], ce, W_cls, b_cls, lossAcc, out);
}
